// SoftCVQLayer_28046136443280
// MI455X (gfx1250) — matrix operand order repaired and checked by reading, not compiled, not run on hardware
//
#include <hip/hip_runtime.h>
#include <hip/hip_bf16.h>
#include <stdint.h>

// ---------------------------------------------------------------------------
// SoftCVQ forward for MI455X (gfx1250, wave32, WMMA + TDM).
// - All big matmuls: bf16 WMMA (v_wmma_f32_16x16x32_bf16), f32 accumulate.
// - LDS tiles are ping-pong double buffered; strided 2D tiles are staged by
//   the Tensor Data Mover (tensor_load_to_lds, TENSORcnt) where available,
//   with LDS padding expressed via the D# pad fields.
// - hc (8192x8192) and the embed Gram matrix are fused into argmax / rowmax
//   epilogues and never materialized.
// ---------------------------------------------------------------------------

typedef __attribute__((ext_vector_type(16))) __bf16       v16bf;
typedef __attribute__((ext_vector_type(8)))  float        v8f;
typedef __attribute__((ext_vector_type(8)))  unsigned int v8u;
typedef __attribute__((ext_vector_type(4)))  unsigned int u32x4;
typedef __attribute__((ext_vector_type(8)))  int          i32x8;
typedef __attribute__((ext_vector_type(4)))  int          i32x4;

#define LDK   40   // LDS row stride (bf16) for 32-wide K tiles: 64B + 16B pad
#define LDK2  72   // LDS row stride (bf16) for 64-wide tiles: 128B + 16B pad

#if defined(__has_builtin)
#if __has_builtin(__builtin_amdgcn_tensor_load_to_lds) && \
    __has_builtin(__builtin_amdgcn_s_wait_tensorcnt)
#define USE_TDM 1
#endif
#endif
#ifndef USE_TDM
#define USE_TDM 0
#endif

__device__ __forceinline__ unsigned short f2bf(float f) {
  unsigned u = __builtin_bit_cast(unsigned, f);
  return (unsigned short)((u + 0x7fffu + ((u >> 16) & 1u)) >> 16);  // RNE
}

__device__ __forceinline__ unsigned hash32(unsigned x) {
  x ^= x >> 16; x *= 0x7feb352du;
  x ^= x >> 15; x *= 0x846ca68bu;
  x ^= x >> 16; return x;
}

#if USE_TDM
// Issue one TDM 2D-tile load: tile_w bf16 elems x tile_h rows from a row-major
// tensor (row stride = stride_elems) into LDS, with (pad_amount+1) DWORDs of LDS
// padding inserted after every 2^(pad_interval+1) DWORDs (our row padding).
__device__ __forceinline__ void tdm_load_tile_bf16(
    unsigned short* lds, const unsigned short* gsrc, unsigned tile_w,
    unsigned tile_h, unsigned stride_elems, unsigned tensor_h,
    unsigned pad_interval, unsigned pad_amount) {
  unsigned long long ga = (unsigned long long)(size_t)gsrc;
  unsigned la = (unsigned)(size_t)lds;  // low 32 bits of generic = LDS offset
  u32x4 g0;
  g0[0] = 1u;                                   // count=1 (valid), user mode
  g0[1] = la;                                   // lds_addr
  g0[2] = (unsigned)ga;                         // global_addr[31:0]
  g0[3] = (unsigned)((ga >> 32) & 0x01FFFFFFu)  // global_addr[56:32]
          | 0x80000000u;                        // type = 2 ("image")
  i32x8 g1;
  g1[0] = (int)((1u << 16)                      // data_size = 1 (2 bytes)
                | (1u << 20)                    // pad_enable
                | (pad_interval << 22) | (pad_amount << 25));
  g1[1] = (int)((stride_elems & 0xFFFFu) << 16);            // tensor_dim0 lo
  g1[2] = (int)(((stride_elems >> 16) & 0xFFFFu) |
                ((tensor_h & 0xFFFFu) << 16));              // dim0 hi | dim1 lo
  g1[3] = (int)(((tensor_h >> 16) & 0xFFFFu) |
                ((tile_w & 0xFFFFu) << 16));                // dim1 hi | tile0
  g1[4] = (int)(tile_h & 0xFFFFu);                          // tile1, tile2=0
  g1[5] = (int)stride_elems;                                // dim0_stride lo32
  g1[6] = 0;                                                // stride hi, s1 lo
  g1[7] = 0;
  i32x4 z4 = {0, 0, 0, 0};
  i32x8 z8 = {0, 0, 0, 0, 0, 0, 0, 0};
  __builtin_amdgcn_tensor_load_to_lds(g0, g1, z4, z4, z8, 0);
}
#endif

// A fragment: 16x32 bf16, M = lane%16, K striped per ISA table:
//   k = (v>=4 ? 16:0) + (lane/16)*8 + (v%4)*2 + e
__device__ __forceinline__ v16bf load_afrag(const unsigned short* At, int ld,
                                            int mbase, int k0) {
  const int l = threadIdx.x & 31;
  const int half = l >> 4;
  const int row = (mbase + (l & 15)) * ld + k0;
  v8u u;
#pragma unroll
  for (int v = 0; v < 8; ++v) {
    int k = ((v & 4) << 2) + (half << 3) + ((v & 3) << 1);
    u[v] = *(const unsigned*)(At + row + k);
  }
  return __builtin_bit_cast(v16bf, u);
}

// B fragment: 32x16 bf16, N = lane%16, K striped same as A fragment.
// Bt is stored transposed in LDS: Bt[n*ld + k].
__device__ __forceinline__ v16bf load_bfrag(const unsigned short* Bt, int ld,
                                            int nbase, int k0) {
  const int l = threadIdx.x & 31;
  const int half = l >> 4;
  const int row = (nbase + (l & 15)) * ld + k0;
  v8u u;
#pragma unroll
  for (int v = 0; v < 8; ++v)
    u[v] = *(const unsigned*)(Bt + row + ((v & 4) << 2) + (half << 3) + ((v & 3) << 1));
  return __builtin_bit_cast(v16bf, u);
}

__device__ __forceinline__ v8f wmma_bf16(v16bf a, v16bf b, v8f c) {
  return __builtin_amdgcn_wmma_f32_16x16x32_bf16(false, a, false, b,
                                                 (short)0, c, false, false);
}

// ---------------------------------------------------------------------------
// Generic C = A(bf16)[M,K] @ B(bf16)[K,N] + bias, f32 out.
// Block tile 128x64, 8 waves (4x2), wave tile 32x32 = 2x2 WMMA tiles.
// Double-buffered LDS; A tile via TDM, B tile transposed manually.
// M=8192, N in {64,512,1024}, K in {64,512,1024} -> no bounds checks.
// ---------------------------------------------------------------------------
__global__ __launch_bounds__(256) void gemm_bf16(
    const unsigned short* __restrict__ A, int lda,
    const unsigned short* __restrict__ B, int ldb,
    const float* __restrict__ bias,
    float* __restrict__ C, int ldc, int K) {
  __shared__ unsigned short At[2][128 * LDK];
  __shared__ unsigned short Bt[2][64 * LDK];
  const int tid = threadIdx.x, wave = tid >> 5, lane = tid & 31;
  const int wm = wave >> 1, wn = wave & 1;
  const int m0 = blockIdx.x * 128, n0 = blockIdx.y * 64;

  auto stageA = [&](int k0, int buf) {
#if USE_TDM
    if (wave == 0)
      tdm_load_tile_bf16(At[buf], A + (size_t)m0 * lda + k0, 32, 128,
                         (unsigned)lda, 8192u, 3u, 3u);
#else
#pragma unroll
    for (int i = 0; i < 2; ++i) {
      int idx = tid + i * 256;
      int r = idx >> 2, c8 = (idx & 3) << 3;
      *(uint4*)(At[buf] + r * LDK + c8) =
          *(const uint4*)(A + (size_t)(m0 + r) * lda + k0 + c8);
    }
#endif
  };
  auto stageB = [&](int k0, int buf) {
    int k = tid >> 3, n8 = (tid & 7) << 3;
    uint4 d = *(const uint4*)(B + (size_t)(k0 + k) * ldb + n0 + n8);
    const unsigned short* s = (const unsigned short*)&d;
#pragma unroll
    for (int j = 0; j < 8; ++j) Bt[buf][(n8 + j) * LDK + k] = s[j];
  };

  v8f acc[2][2] = {};
  stageA(0, 0);
  stageB(0, 0);
  const int nk = K >> 5;
  for (int ki = 0; ki < nk; ++ki) {
    const int buf = ki & 1;
#if USE_TDM
    if (wave == 0) __builtin_amdgcn_s_wait_tensorcnt(0);
#endif
    __syncthreads();
    if (ki + 1 < nk) {  // stage next tile while computing this one
      stageA((ki + 1) << 5, buf ^ 1);
      stageB((ki + 1) << 5, buf ^ 1);
    }
    v16bf a0 = load_afrag(At[buf], LDK, wm * 32, 0);
    v16bf a1 = load_afrag(At[buf], LDK, wm * 32 + 16, 0);
    v16bf b0 = load_bfrag(Bt[buf], LDK, wn * 32, 0);
    v16bf b1 = load_bfrag(Bt[buf], LDK, wn * 32 + 16, 0);
    acc[0][0] = wmma_bf16(a0, b0, acc[0][0]);
    acc[0][1] = wmma_bf16(a0, b1, acc[0][1]);
    acc[1][0] = wmma_bf16(a1, b0, acc[1][0]);
    acc[1][1] = wmma_bf16(a1, b1, acc[1][1]);
  }

#pragma unroll
  for (int tm = 0; tm < 2; ++tm)
#pragma unroll
    for (int tn = 0; tn < 2; ++tn) {
      int n = n0 + wn * 32 + tn * 16 + (lane & 15);
      float bv = bias[n];
#pragma unroll
      for (int g = 0; g < 8; ++g) {
        int m = m0 + wm * 32 + tm * 16 + ((lane >> 4) << 3) + g;
        C[(size_t)m * ldc + n] = acc[tm][tn][g] + bv;
      }
    }
}

// ---------------------------------------------------------------------------
// hc = H[8192,64] @ E^T, fused 2*hc + gumbel argmax per row (never stored).
// One block = 128 rows of H; loops over all 8192 codes in 64-wide tiles.
// Embed rows are already B^T layout -> TDM-stageable without transpose.
// ---------------------------------------------------------------------------
__global__ __launch_bounds__(256) void hc_argmax(
    const unsigned short* __restrict__ Hb,  // [8192,64] bf16, normalized
    const unsigned short* __restrict__ Eb,  // [8192,64] bf16, normalized
    const int* __restrict__ mask,           // [8192]
    int* __restrict__ code,                 // [8192] raw sampled codes
    float* __restrict__ vq_code_out)        // [8192] masked, as float
{
  __shared__ unsigned short At[128 * LDK2];
  __shared__ unsigned short Bt[2][64 * LDK2];
  __shared__ float redv[4][32][2];
  __shared__ int   redi[4][32][2];
  const int tid = threadIdx.x, wave = tid >> 5, lane = tid & 31;
  const int wm = wave >> 1, wn = wave & 1;
  const int m0 = blockIdx.x * 128;

  auto stageB = [&](int nb, int buf) {
#if USE_TDM
    if (wave == 0)
      tdm_load_tile_bf16(Bt[buf], Eb + (size_t)nb * 64, 64, 64, 64u, 8192u,
                         4u, 3u);
#else
#pragma unroll
    for (int i = 0; i < 2; ++i) {
      int idx = tid + i * 256;
      int r = idx >> 3, c8 = (idx & 7) << 3;
      *(uint4*)(Bt[buf] + r * LDK2 + c8) =
          *(const uint4*)(Eb + (nb + r) * 64 + c8);
    }
#endif
  };
#if USE_TDM
  if (wave == 0)
    tdm_load_tile_bf16(At, Hb + (size_t)m0 * 64, 64, 128, 64u, 8192u, 4u, 3u);
#else
#pragma unroll
  for (int i = 0; i < 4; ++i) {
    int idx = tid + i * 256;
    int r = idx >> 3, c8 = (idx & 7) << 3;
    *(uint4*)(At + r * LDK2 + c8) = *(const uint4*)(Hb + (m0 + r) * 64 + c8);
  }
#endif
  stageB(0, 0);

  float bestv[2][8];
  int   besti[2][8];
#pragma unroll
  for (int tm = 0; tm < 2; ++tm)
#pragma unroll
    for (int g = 0; g < 8; ++g) { bestv[tm][g] = -1e30f; besti[tm][g] = 0; }

  for (int it = 0; it < 128; ++it) {
    const int buf = it & 1;
    const int nbase = it * 64;
#if USE_TDM
    if (wave == 0) __builtin_amdgcn_s_wait_tensorcnt(0);
#endif
    __syncthreads();
    if (it + 1 < 128) stageB(nbase + 64, buf ^ 1);

    v8f acc[2][2] = {};
#pragma unroll
    for (int ks = 0; ks < 2; ++ks) {
      v16bf a0 = load_afrag(At, LDK2, wm * 32, ks * 32);
      v16bf a1 = load_afrag(At, LDK2, wm * 32 + 16, ks * 32);
      v16bf b0 = load_bfrag(Bt[buf], LDK2, wn * 32, ks * 32);
      v16bf b1 = load_bfrag(Bt[buf], LDK2, wn * 32 + 16, ks * 32);
      acc[0][0] = wmma_bf16(a0, b0, acc[0][0]);
      acc[0][1] = wmma_bf16(a0, b1, acc[0][1]);
      acc[1][0] = wmma_bf16(a1, b0, acc[1][0]);
      acc[1][1] = wmma_bf16(a1, b1, acc[1][1]);
    }
#pragma unroll
    for (int tm = 0; tm < 2; ++tm)
#pragma unroll
      for (int tn = 0; tn < 2; ++tn)
#pragma unroll
        for (int g = 0; g < 8; ++g) {
          int n = nbase + wn * 32 + tn * 16 + (lane & 15);
          int m = m0 + wm * 32 + tm * 16 + ((lane >> 4) << 3) + g;
          unsigned h = hash32((unsigned)(m * 8192 + n));
          float u = ((float)(h >> 8) + 0.5f) * (1.0f / 16777216.0f);
          float gum = -__logf(-__logf(u));
          float v = 2.0f * acc[tm][tn][g] + gum;
          if (v > bestv[tm][g] ||
              (v == bestv[tm][g] && n < besti[tm][g])) {
            bestv[tm][g] = v; besti[tm][g] = n;
          }
        }
  }

  // reduce across the 16 lanes holding one row, then across wn pairs via LDS
#pragma unroll
  for (int tm = 0; tm < 2; ++tm)
#pragma unroll
    for (int g = 0; g < 8; ++g) {
      float v = bestv[tm][g]; int b = besti[tm][g];
#pragma unroll
      for (int s = 1; s < 16; s <<= 1) {
        float ov = __shfl_xor(v, s, 32);
        int   ob = __shfl_xor(b, s, 32);
        if (ov > v || (ov == v && ob < b)) { v = ov; b = ob; }
      }
      if ((lane & 15) == 0) {
        int ri = tm * 16 + ((lane >> 4) << 3) + g;
        redv[wm][ri][wn] = v; redi[wm][ri][wn] = b;
      }
    }
  __syncthreads();
  if (tid < 128) {
    int wmx = tid >> 5, ri = tid & 31;
    float v0 = redv[wmx][ri][0], v1 = redv[wmx][ri][1];
    int   b0 = redi[wmx][ri][0], b1 = redi[wmx][ri][1];
    bool take1 = (v1 > v0) || (v1 == v0 && b1 < b0);
    int b = take1 ? b1 : b0;
    int m = m0 + wmx * 32 + ri;
    code[m] = b;
    vq_code_out[m] = (float)(mask[m] ? b : 0);
  }
}

// ---------------------------------------------------------------------------
// vq_loss: row-max of embed @ embed^T with diagonal forced to -1; per-block
// row-max sums written to blocksum[64] (deterministic final reduce).
// ---------------------------------------------------------------------------
__global__ __launch_bounds__(256) void sim_rowmax(
    const unsigned short* __restrict__ Eb, float* __restrict__ blocksum) {
  __shared__ unsigned short At[128 * LDK2];
  __shared__ unsigned short Bt[2][64 * LDK2];
  __shared__ float redv[4][32][2];
  __shared__ float rs[128];
  const int tid = threadIdx.x, wave = tid >> 5, lane = tid & 31;
  const int wm = wave >> 1, wn = wave & 1;
  const int m0 = blockIdx.x * 128;

  auto stageB = [&](int nb, int buf) {
#if USE_TDM
    if (wave == 0)
      tdm_load_tile_bf16(Bt[buf], Eb + (size_t)nb * 64, 64, 64, 64u, 8192u,
                         4u, 3u);
#else
#pragma unroll
    for (int i = 0; i < 2; ++i) {
      int idx = tid + i * 256;
      int r = idx >> 3, c8 = (idx & 7) << 3;
      *(uint4*)(Bt[buf] + r * LDK2 + c8) =
          *(const uint4*)(Eb + (nb + r) * 64 + c8);
    }
#endif
  };
#if USE_TDM
  if (wave == 0)
    tdm_load_tile_bf16(At, Eb + (size_t)m0 * 64, 64, 128, 64u, 8192u, 4u, 3u);
#else
#pragma unroll
  for (int i = 0; i < 4; ++i) {
    int idx = tid + i * 256;
    int r = idx >> 3, c8 = (idx & 7) << 3;
    *(uint4*)(At + r * LDK2 + c8) = *(const uint4*)(Eb + (m0 + r) * 64 + c8);
  }
#endif
  stageB(0, 0);

  float best[2][8];
#pragma unroll
  for (int tm = 0; tm < 2; ++tm)
#pragma unroll
    for (int g = 0; g < 8; ++g) best[tm][g] = -1e30f;

  for (int it = 0; it < 128; ++it) {
    const int buf = it & 1;
    const int nbase = it * 64;
#if USE_TDM
    if (wave == 0) __builtin_amdgcn_s_wait_tensorcnt(0);
#endif
    __syncthreads();
    if (it + 1 < 128) stageB(nbase + 64, buf ^ 1);

    v8f acc[2][2] = {};
#pragma unroll
    for (int ks = 0; ks < 2; ++ks) {
      v16bf a0 = load_afrag(At, LDK2, wm * 32, ks * 32);
      v16bf a1 = load_afrag(At, LDK2, wm * 32 + 16, ks * 32);
      v16bf b0 = load_bfrag(Bt[buf], LDK2, wn * 32, ks * 32);
      v16bf b1 = load_bfrag(Bt[buf], LDK2, wn * 32 + 16, ks * 32);
      acc[0][0] = wmma_bf16(a0, b0, acc[0][0]);
      acc[0][1] = wmma_bf16(a0, b1, acc[0][1]);
      acc[1][0] = wmma_bf16(a1, b0, acc[1][0]);
      acc[1][1] = wmma_bf16(a1, b1, acc[1][1]);
    }
#pragma unroll
    for (int tm = 0; tm < 2; ++tm)
#pragma unroll
      for (int tn = 0; tn < 2; ++tn)
#pragma unroll
        for (int g = 0; g < 8; ++g) {
          int n = nbase + wn * 32 + tn * 16 + (lane & 15);
          int m = m0 + wm * 32 + tm * 16 + ((lane >> 4) << 3) + g;
          float v = (m == n) ? -1.0f : acc[tm][tn][g];
          if (v > best[tm][g]) best[tm][g] = v;
        }
  }

#pragma unroll
  for (int tm = 0; tm < 2; ++tm)
#pragma unroll
    for (int g = 0; g < 8; ++g) {
      float v = best[tm][g];
#pragma unroll
      for (int s = 1; s < 16; s <<= 1) {
        float ov = __shfl_xor(v, s, 32);
        if (ov > v) v = ov;
      }
      if ((lane & 15) == 0) {
        int ri = tm * 16 + ((lane >> 4) << 3) + g;
        redv[wm][ri][wn] = v;
      }
    }
  __syncthreads();
  if (tid < 128) {
    int wmx = tid >> 5, ri = tid & 31;
    float a = redv[wmx][ri][0], b = redv[wmx][ri][1];
    rs[tid] = a > b ? a : b;
  }
  __syncthreads();
  if (tid == 0) {
    float s = 0.f;
    for (int i = 0; i < 128; ++i) s += rs[i];
    blocksum[blockIdx.x] = s;
  }
}

// --------------------------- small helper kernels ---------------------------

__global__ void f32_to_bf16(const float* __restrict__ in,
                            unsigned short* __restrict__ out, int n) {
  int i = blockIdx.x * blockDim.x + threadIdx.x;
  if (i < n) out[i] = f2bf(in[i]);
}

// x0 = bits @ w_in + b_in  (bits generated from the row index, K=13)
__global__ void codebook_in(const float* __restrict__ w_in,
                            const float* __restrict__ b_in,
                            float* __restrict__ y) {
  int i = blockIdx.x * blockDim.x + threadIdx.x;
  if (i >= 8192 * 1024) return;
  int r = i >> 10, c = i & 1023;
  float s = b_in[c];
#pragma unroll
  for (int j = 0; j < 13; ++j)
    if ((r >> (12 - j)) & 1) s += w_in[j * 1024 + c];
  y[i] = s;
}

// two-pass BN stats (deterministic): per-(rowgroup, column) partial sums
__global__ void col_stats_partial(const float* __restrict__ y,
                                  float* __restrict__ part) {
  int rg = blockIdx.x >> 2, cg = blockIdx.x & 3;
  int c = cg * 256 + threadIdx.x;
  float s = 0.f, s2 = 0.f;
  for (int r = rg * 1024; r < rg * 1024 + 1024; ++r) {
    float x = y[(size_t)r * 1024 + c];
    s += x; s2 += x * x;
  }
  part[rg * 1024 + c] = s;
  part[8192 + rg * 1024 + c] = s2;
}

__global__ void col_stats_final(const float* __restrict__ part,
                                const float* __restrict__ gamma,
                                const float* __restrict__ beta,
                                float* __restrict__ ss) {
  int c = blockIdx.x * blockDim.x + threadIdx.x;
  if (c >= 1024) return;
  float s = 0.f, s2 = 0.f;
  for (int rg = 0; rg < 8; ++rg) {
    s += part[rg * 1024 + c];
    s2 += part[8192 + rg * 1024 + c];
  }
  float mean = s * (1.f / 8192.f);
  float var = s2 * (1.f / 8192.f) - mean * mean;
  float sc = gamma[c] * rsqrtf(var + 1e-5f);
  ss[c] = sc;
  ss[1024 + c] = beta[c] - mean * sc;
}

__global__ void bn_relu_bf16(const float* __restrict__ y,
                             const float* __restrict__ ss,
                             unsigned short* __restrict__ xb) {
  int i = blockIdx.x * blockDim.x + threadIdx.x;
  if (i >= 8192 * 1024) return;
  int c = i & 1023;
  float v = ss[c] * y[i] + ss[1024 + c];
  xb[i] = f2bf(v > 0.f ? v : 0.f);
}

// row L2-normalize 8192x64 f32 -> bf16 (one wave per row)
__global__ void normalize_rows(const float* __restrict__ in,
                               unsigned short* __restrict__ out) {
  int warp = (blockIdx.x * blockDim.x + threadIdx.x) >> 5;
  int lane = threadIdx.x & 31;
  if (warp >= 8192) return;
  const float* r = in + (size_t)warp * 64;
  float x0 = r[lane], x1 = r[lane + 32];
  float s = x0 * x0 + x1 * x1;
#pragma unroll
  for (int sh = 16; sh >= 1; sh >>= 1) s += __shfl_xor(s, sh, 32);
  float inv = 1.f / (sqrtf(s) + 1e-6f);
  out[(size_t)warp * 64 + lane] = f2bf(x0 * inv);
  out[(size_t)warp * 64 + lane + 32] = f2bf(x1 * inv);
}

__global__ void gather_qrows(const unsigned short* __restrict__ Eb,
                             const int* __restrict__ code,
                             const int* __restrict__ mask,
                             unsigned short* __restrict__ q) {
  int i = blockIdx.x * blockDim.x + threadIdx.x;
  if (i >= 8192 * 64) return;
  int m = i >> 6, k = i & 63;
  q[i] = mask[m] ? Eb[code[m] * 64 + k] : (unsigned short)0;
}

__global__ void loss_final(const float* __restrict__ blocksum,
                           float* __restrict__ out) {
  if (threadIdx.x == 0) {
    float s = 0.f;
    for (int i = 0; i < 64; ++i) s += blocksum[i];
    *out = s * (1.f / 8192.f);
  }
}

// ---------------------------------------------------------------------------

extern "C" void kernel_launch(void* const* d_in, const int* in_sizes, int n_in,
                              void* d_out, int out_size, void* d_ws,
                              size_t ws_size, hipStream_t stream) {
  (void)in_sizes; (void)n_in; (void)out_size; (void)ws_size;
  const float* h_in   = (const float*)d_in[0];
  const int*   mask   = (const int*)  d_in[1];
  const float* proj_w = (const float*)d_in[2];
  const float* proj_b = (const float*)d_in[3];
  const float* inv_w  = (const float*)d_in[4];
  const float* inv_b  = (const float*)d_in[5];
  const float* w_in   = (const float*)d_in[6];
  const float* b_in   = (const float*)d_in[7];
  const float* wh     = (const float*)d_in[8];
  const float* bh     = (const float*)d_in[9];
  const float* gammas = (const float*)d_in[10];
  const float* betas  = (const float*)d_in[11];
  const float* w_out  = (const float*)d_in[12];
  const float* b_out  = (const float*)d_in[13];
  float* out = (float*)d_out;  // [quantized 4194304][vq_code 8192][vq_loss 1]

  char* w = (char*)d_ws;
  auto alloc = [&](size_t bytes) {
    char* p = w;
    w += (bytes + 255) & ~(size_t)255;
    return p;
  };
  float*          y      = (float*)alloc(8192ull * 1024 * 4);  // activations
  unsigned short* xbA    = (unsigned short*)alloc(8192ull * 1024 * 2);
  unsigned short* xbB    = (unsigned short*)alloc(8192ull * 1024 * 2);
  unsigned short* whb    = (unsigned short*)alloc(4ull * 1024 * 1024 * 2);
  unsigned short* hinb   = (unsigned short*)alloc(8192ull * 512 * 2);
  unsigned short* projwb = (unsigned short*)alloc(512ull * 64 * 2);
  unsigned short* invwb  = (unsigned short*)alloc(64ull * 512 * 2);
  unsigned short* woutb  = (unsigned short*)alloc(1024ull * 64 * 2);
  unsigned short* embb   = (unsigned short*)alloc(8192ull * 64 * 2);
  unsigned short* hb     = (unsigned short*)alloc(8192ull * 64 * 2);
  unsigned short* qrows  = (unsigned short*)alloc(8192ull * 64 * 2);
  float*          part   = (float*)alloc(2ull * 8 * 1024 * 4);
  float*          ss     = (float*)alloc(2048ull * 4);
  int*            code   = (int*)alloc(8192ull * 4);
  float*          bsum   = (float*)alloc(64ull * 4);
  // y is free once the last bn_relu has consumed it -> reuse for 64-wide f32
  float* embf = y;                 // [8192,64] f32 (pre-normalize)
  float* hf   = y + 8192ull * 64;  // [8192,64] f32 (pre-normalize)

  // weight / input conversions to bf16
  f32_to_bf16<<<16384, 256, 0, stream>>>(wh, whb, 4 * 1024 * 1024);
  f32_to_bf16<<<16384, 256, 0, stream>>>(h_in, hinb, 8192 * 512);
  f32_to_bf16<<<128, 256, 0, stream>>>(proj_w, projwb, 512 * 64);
  f32_to_bf16<<<128, 256, 0, stream>>>(inv_w, invwb, 64 * 512);
  f32_to_bf16<<<256, 256, 0, stream>>>(w_out, woutb, 1024 * 64);

  // codebook MLP: x0 then 5x (BN+ReLU -> GEMM)
  codebook_in<<<32768, 256, 0, stream>>>(w_in, b_in, y);
  for (int l = 0; l < 5; ++l) {
    col_stats_partial<<<32, 256, 0, stream>>>(y, part);
    col_stats_final<<<4, 256, 0, stream>>>(part, gammas + l * 1024,
                                           betas + l * 1024, ss);
    unsigned short* xb = (l & 1) ? xbB : xbA;
    bn_relu_bf16<<<32768, 256, 0, stream>>>(y, ss, xb);
    if (l < 4)
      gemm_bf16<<<dim3(64, 16), 256, 0, stream>>>(
          xb, 1024, whb + (size_t)l * 1024 * 1024, 1024, bh + l * 1024,
          y, 1024, 1024);
    else
      gemm_bf16<<<dim3(64, 1), 256, 0, stream>>>(xb, 1024, woutb, 64, b_out,
                                                 embf, 64, 1024);
  }
  normalize_rows<<<1024, 256, 0, stream>>>(embf, embb);

  // h = normalize(h_in @ proj_w + proj_b)
  gemm_bf16<<<dim3(64, 1), 256, 0, stream>>>(hinb, 512, projwb, 64, proj_b,
                                             hf, 64, 512);
  normalize_rows<<<1024, 256, 0, stream>>>(hf, hb);

  // categorical sample (Gumbel-argmax over 2*hc) + vq_code output
  hc_argmax<<<64, 256, 0, stream>>>(hb, embb, mask, code, out + 4194304);

  // vq_loss
  sim_rowmax<<<64, 256, 0, stream>>>(embb, bsum);
  loss_final<<<1, 64, 0, stream>>>(bsum, out + 4194304 + 8192);

  // quantized = (embed[code] * mask) @ inv_w + inv_b
  gather_qrows<<<2048, 256, 0, stream>>>(embb, code, mask, qrows);
  gemm_bf16<<<dim3(64, 8), 256, 0, stream>>>(qrows, 64, invwb, 512, inv_b,
                                             out, 512, 64);
}
